// GraphViTModel_3693671874574
// MI455X (gfx1250) — matrix operand order repaired and checked by reading, not compiled, not run on hardware
//
#include <hip/hip_runtime.h>
#include <hip/hip_bf16.h>
#include <math.h>

// ---------------------------------------------------------------------------
// GraphViT forward for MI455X (gfx1250, wave32, WMMA).
// GEMMs: v_wmma_f32_16x16x32_f16 via LDS-staged f16 tiles (ds_load_b128
// fragment reads) + global_prefetch of the next k-tile.
// Attention: fused flash-style; QKV pre-converted to f16 (V pre-transposed
// per head) so K/V chunks are staged with GLOBAL_LOAD_ASYNC_TO_LDS_B128
// (ASYNCcnt) and ALL fragments are contiguous ds_load_b128 reads.
// Cross-lane reductions use ds_swizzle_b32 immediates (no bpermute setup).
// ---------------------------------------------------------------------------

#define BSZ    2
#define NN     3072
#define NEDGE  49152
#define ENCD   128
#define NHEAD  4
#define HD     32
#define DT_STEP 0.05f
#define LN_EPS_ 1e-5f
#define PI_F 3.14159265358979f

typedef __attribute__((ext_vector_type(16))) _Float16 v16h;
typedef __attribute__((ext_vector_type(8)))  _Float16 v8h;
typedef __attribute__((ext_vector_type(4)))  _Float16 v4h;
typedef __attribute__((ext_vector_type(8)))  float    v8f;

__device__ __forceinline__ float siluf(float x){ return x / (1.0f + __expf(-x)); }

// ds_swizzle_b32 butterfly exchange: group-of-32 mode, xor_mask<<10 | and=0x1f
template<int MASK>
__device__ __forceinline__ float swz_xor(float v){
  return __int_as_float(
      __builtin_amdgcn_ds_swizzle(__float_as_int(v), 0x1f | (MASK << 10)));
}
__device__ __forceinline__ float wave_sum32(float v){
  v += swz_xor<1>(v); v += swz_xor<2>(v); v += swz_xor<4>(v);
  v += swz_xor<8>(v); v += swz_xor<16>(v);
  return v;
}
__device__ __forceinline__ float group16_sum(float v){
  v += swz_xor<1>(v); v += swz_xor<2>(v); v += swz_xor<4>(v); v += swz_xor<8>(v);
  return v;
}
__device__ __forceinline__ float group16_max(float v){
  v = fmaxf(v, swz_xor<1>(v)); v = fmaxf(v, swz_xor<2>(v));
  v = fmaxf(v, swz_xor<4>(v)); v = fmaxf(v, swz_xor<8>(v));
  return v;
}

__device__ __forceinline__ v8f wmma_f16(v16h a, v16h b, v8f c){
  return __builtin_amdgcn_wmma_f32_16x16x32_f16(false, a, false, b, (short)0, c,
                                                false, false);
}
__device__ __forceinline__ v16h cat8(v8h lo, v8h hi){
  return __builtin_shufflevector(lo, hi, 0,1,2,3,4,5,6,7,8,9,10,11,12,13,14,15);
}
__device__ __forceinline__ v4h cvt4(float4 f){
  v4h h; h[0]=(_Float16)f.x; h[1]=(_Float16)f.y; h[2]=(_Float16)f.z; h[3]=(_Float16)f.w;
  return h;
}
// 16-byte async global->LDS copy (tracked by ASYNCcnt).
__device__ __forceinline__ void async_copy16(const _Float16* src, void* ldsDst){
  uint32_t dst = (uint32_t)(uintptr_t)ldsDst;   // low 32 bits of flat = LDS offset
  asm volatile("global_load_async_to_lds_b128 %0, %1, off"
               :: "v"(dst), "v"(src) : "memory");
}
__device__ __forceinline__ void async_wait0(){
  asm volatile("s_wait_asynccnt 0x0" ::: "memory");
}

// ---------------------------------------------------------------------------
// WMMA GEMM:  Cout = act( gather(A) @ W + bias + Cin )
// Host guarantees: K % 32 == 0, M % 128 == 0, lda % 4 == 0, ldw % 4 == 0.
// Tile 128x64x32, 256 threads (8 waves), wave -> 32x32 of C (4 WMMA/k-step).
// LDS strides 48 halves (96B) keep fragment reads as ds_load_b128.
// ---------------------------------------------------------------------------
#define GBM 128
#define GBN 64
#define GBK 32
#define LDH 48

template <bool GATHER>
__global__ __launch_bounds__(256)
void gemm_wmma_kernel(const float* __restrict__ A, int lda,
                      const int* __restrict__ rowIdx,
                      const float* __restrict__ W, int ldw,
                      const float* __restrict__ bias,
                      const float* __restrict__ Cin,
                      float* __restrict__ Cout,
                      int M, int K, int Nc, int act)
{
  __shared__ _Float16 As[GBM][LDH];   // [row][k]
  __shared__ _Float16 Bst[GBN][LDH];  // [n][k]  (transposed weight tile)
  __shared__ int ridxS[GBM];

  const int tid  = threadIdx.x;
  const int lane = tid & 31;
  const int wave = tid >> 5;
  const int wm   = wave >> 1;   // 0..3
  const int wn   = wave & 1;    // 0..1
  const int lh   = lane >> 4;
  const int ll   = lane & 15;

  const int m_blk = blockIdx.x * GBM;
  const int n_blk = blockIdx.y * GBN;
  const bool n_full = (n_blk + GBN) <= Nc;

  if (GATHER){
    if (tid < GBM) ridxS[tid] = rowIdx[m_blk + tid];
    __syncthreads();
  }

  v8f zero = {};
  v8f acc[2][2];
  acc[0][0] = zero; acc[0][1] = zero; acc[1][0] = zero; acc[1][1] = zero;

  for (int k0 = 0; k0 < K; k0 += GBK){
    // prefetch next k-tile into cache while this one is staged/computed
    if (k0 + GBK < K){
      int pr = tid >> 1;                       // 128 rows, 2 threads each
      int prow = GATHER ? ridxS[pr] : (m_blk + pr);
      __builtin_prefetch(A + (size_t)prow * lda + k0 + GBK + (tid & 1)*16, 0, 1);
      __builtin_prefetch(W + (size_t)(k0 + GBK + (tid & 31)) * ldw + n_blk, 0, 1);
    }
    // ---- stage A tile: 1024 float4, 4 per thread ----
#pragma unroll
    for (int i = 0; i < 4; ++i){
      int e  = i*256 + tid;
      int r  = e >> 3;
      int c4 = e & 7;
      int row = GATHER ? ridxS[r] : (m_blk + r);
      float4 f = *(const float4*)(A + (size_t)row * lda + k0 + c4*4);
      *(v4h*)&As[r][c4*4] = cvt4(f);
    }
    // ---- stage W tile transposed: 512 float4, 2 per thread ----
#pragma unroll
    for (int i = 0; i < 2; ++i){
      int e  = i*256 + tid;
      int r  = e >> 4;          // k row 0..31
      int c4 = e & 15;          // n float4 0..15
      v4h h;
      if (n_full){
        float4 f = *(const float4*)(W + (size_t)(k0 + r) * ldw + n_blk + c4*4);
        h = cvt4(f);
      } else {
#pragma unroll
        for (int j = 0; j < 4; ++j){
          int n = n_blk + c4*4 + j;
          h[j] = (_Float16)((n < Nc) ? W[(size_t)(k0 + r) * ldw + n] : 0.0f);
        }
      }
#pragma unroll
      for (int j = 0; j < 4; ++j) Bst[c4*4 + j][r] = h[j];
    }
    __syncthreads();

    // ---- fragments: contiguous 16B LDS reads ----
    v16h af[2], bf[2];
#pragma unroll
    for (int tm = 0; tm < 2; ++tm){
      int mb = wm*32 + tm*16 + ll;
      af[tm] = cat8(*(const v8h*)&As[mb][8*lh],
                    *(const v8h*)&As[mb][16 + 8*lh]);
    }
#pragma unroll
    for (int tn = 0; tn < 2; ++tn){
      int nb = wn*32 + tn*16 + ll;
      bf[tn] = cat8(*(const v8h*)&Bst[nb][8*lh],
                    *(const v8h*)&Bst[nb][16 + 8*lh]);
    }
#pragma unroll
    for (int tm = 0; tm < 2; ++tm)
#pragma unroll
      for (int tn = 0; tn < 2; ++tn)
        acc[tm][tn] = wmma_f16(af[tm], bf[tn], acc[tm][tn]);
    __syncthreads();
  }

  // ---- epilogue: bias / residual / activation ----
#pragma unroll
  for (int tm = 0; tm < 2; ++tm){
#pragma unroll
    for (int tn = 0; tn < 2; ++tn){
#pragma unroll
      for (int r = 0; r < 8; ++r){
        int m = m_blk + wm*32 + tm*16 + r + 8*lh;
        int n = n_blk + wn*32 + tn*16 + ll;
        if (n < Nc){
          float v = acc[tm][tn][r];
          if (bias) v += bias[n];
          if (Cin)  v += Cin[(size_t)m * Nc + n];
          if (act == 1)      v = siluf(v);
          else if (act == 2) v = fmaxf(v, 0.0f);
          Cout[(size_t)m * Nc + n] = v;
        }
      }
    }
  }
}

// ---------------------------------------------------------------------------
// Fused flash attention over pre-converted f16 QKV (V pre-transposed).
// Block = 128 threads (4 waves); wave w owns Q rows m0 = bx*64 + w*16 for
// head by, batch bz. 32-token K/V chunks staged to LDS with
// global_load_async_to_lds_b128 (no VALU), shared by all 4 waves.
// qkvh: (B*N) x 384 f16 [q|k|v];  vth: [b][h][hd=32][token=N] f16;
// out: (B*N) x 128 fp32.
// ---------------------------------------------------------------------------
__global__ __launch_bounds__(128)
void flash_attn_kernel(const _Float16* __restrict__ qkvh,
                       const _Float16* __restrict__ vth,
                       float* __restrict__ out)
{
  const int tid  = threadIdx.x;
  const int lane = tid & 31;
  const int wave = tid >> 5;
  const int lh = lane >> 4;
  const int ll = lane & 15;
  const int m0 = blockIdx.x * 64 + wave * 16;
  const int h  = blockIdx.y;
  const int b  = blockIdx.z;
  const float scale = 0.17677669529663687f;  // 1/sqrt(32)

  const _Float16* base  = qkvh + (size_t)b * NN * 384;
  const _Float16* vbase = vth + ((size_t)(b*NHEAD + h)) * HD * NN;

  __shared__ _Float16 Kl[32][32];            // [token][hd]  (async-staged)
  __shared__ _Float16 Vt[32][32];            // [hd][token]  (async-staged)
  __shared__ float    mS[4][16], lS[4][16];
  __shared__ _Float16 pS[4][16][LDH];        // per-wave P tile [m][token]

  // Q fragment (A layout 16x32): two contiguous 16B f16 loads, scale folded
  v16h qf;
  {
    const _Float16* qp = base + (size_t)(m0 + ll)*384 + h*HD;
    v16h qr = cat8(*(const v8h*)(qp + 8*lh), *(const v8h*)(qp + 16 + 8*lh));
#pragma unroll
    for (int e = 0; e < 16; ++e) qf[e] = (_Float16)((float)qr[e] * scale);
  }

  if (lane < 16){ mS[wave][lane] = -1e30f; lS[wave][lane] = 0.0f; }

  const int srow = tid >> 2;   // staging row 0..31 (K: token, V: hd)
  const int sc   = tid & 3;    // 16B chunk within 32 halves

  v8f zero = {};
  v8f o0 = zero, o1 = zero;

  for (int t0 = 0; t0 < NN; t0 += 32){
    // ---- async DMA staging of K and transposed V (one b128 per thread) ----
    {
      const _Float16* ksrc = base + (size_t)(t0 + srow)*384 + ENCD + h*HD + sc*8;
      async_copy16(ksrc, &Kl[srow][sc*8]);
      const _Float16* vsrc = vbase + (size_t)srow*NN + t0 + sc*8;
      async_copy16(vsrc, &Vt[srow][sc*8]);
      async_wait0();
    }
    __syncthreads();

    // ---- scores: Q @ K^T for two 16-token tiles ----
    v16h kf0 = cat8(*(const v8h*)&Kl[ll][8*lh],      *(const v8h*)&Kl[ll][16 + 8*lh]);
    v16h kf1 = cat8(*(const v8h*)&Kl[16 + ll][8*lh], *(const v8h*)&Kl[16 + ll][16 + 8*lh]);
    v8f s0 = wmma_f16(qf, kf0, zero);
    v8f s1 = wmma_f16(qf, kf1, zero);

    // ---- online softmax over this 32-key chunk ----
    float corr[8];
#pragma unroll
    for (int r = 0; r < 8; ++r){
      float mx = group16_max(fmaxf(s0[r], s1[r]));
      float mold = mS[wave][r + 8*lh];
      float mnew = fmaxf(mold, mx);
      float p0 = __expf(s0[r] - mnew);
      float p1 = __expf(s1[r] - mnew);
      s0[r] = p0; s1[r] = p1;
      float rowsum = group16_sum(p0 + p1);
      float c = __expf(mold - mnew);
      corr[r] = c;
      float lnew = lS[wave][r + 8*lh] * c + rowsum;
      if (ll == 0){ mS[wave][r + 8*lh] = mnew; lS[wave][r + 8*lh] = lnew; }
    }
#pragma unroll
    for (int r = 0; r < 8; ++r){ o0[r] *= corr[r]; o1[r] *= corr[r]; }

    // ---- P: C-layout -> LDS -> A-layout fragment ----
#pragma unroll
    for (int r = 0; r < 8; ++r){
      pS[wave][r + 8*lh][ll]      = (_Float16)s0[r];
      pS[wave][r + 8*lh][16 + ll] = (_Float16)s1[r];
    }
    v16h pf = cat8(*(const v8h*)&pS[wave][ll][8*lh],
                   *(const v8h*)&pS[wave][ll][16 + 8*lh]);

    // ---- O += P @ V: V fragments are contiguous reads of transposed V ----
    v16h vf0 = cat8(*(const v8h*)&Vt[ll][8*lh],      *(const v8h*)&Vt[ll][16 + 8*lh]);
    v16h vf1 = cat8(*(const v8h*)&Vt[16 + ll][8*lh], *(const v8h*)&Vt[16 + ll][16 + 8*lh]);
    o0 = wmma_f16(pf, vf0, o0);
    o1 = wmma_f16(pf, vf1, o1);
    __syncthreads();
  }

#pragma unroll
  for (int r = 0; r < 8; ++r){
    float inv = 1.0f / lS[wave][r + 8*lh];
    size_t row = (size_t)(b*NN + m0 + r + 8*lh);
    out[row*ENCD + h*HD + ll]      = o0[r] * inv;
    out[row*ENCD + h*HD + 16 + ll] = o1[r] * inv;
  }
}

// ---------------------------------------------------------------------------
// LayerNorm over last dim (=128): one wave per row, float4 per lane.
// ---------------------------------------------------------------------------
__global__ __launch_bounds__(256)
void ln_kernel(const float* __restrict__ x, const float* __restrict__ g,
               const float* __restrict__ bt, float* __restrict__ y, int rows)
{
  int wave = threadIdx.x >> 5, lane = threadIdx.x & 31;
  int row = blockIdx.x * 8 + wave;
  if (row >= rows) return;
  const float* xr = x + (size_t)row * ENCD;
  float4 v = *(const float4*)(xr + lane*4);
  float s = wave_sum32(v.x + v.y + v.z + v.w);
  float mean = s * (1.0f / 128.0f);
  float dx = v.x-mean, dy = v.y-mean, dz = v.z-mean, dw = v.w-mean;
  float q = wave_sum32(dx*dx + dy*dy + dz*dz + dw*dw);
  float inv = rsqrtf(q * (1.0f / 128.0f) + LN_EPS_);
  float4 gv = *(const float4*)(g + lane*4);
  float4 bv = *(const float4*)(bt + lane*4);
  float4 o;
  o.x = dx*inv*gv.x + bv.x; o.y = dy*inv*gv.y + bv.y;
  o.z = dz*inv*gv.z + bv.z; o.w = dw*inv*gv.w + bv.w;
  *(float4*)(y + (size_t)row * ENCD + lane*4) = o;
}

// --------------------------- small / elementwise ---------------------------

__global__ void f32_to_f16_kernel(const float* __restrict__ x,
                                  _Float16* __restrict__ y, int n4)
{
  int i = blockIdx.x * blockDim.x + threadIdx.x;
  if (i >= n4) return;
  float4 f = *(const float4*)(x + (size_t)i*4);
  *(v4h*)(y + (size_t)i*4) = cvt4(f);
}

// V part of QKV -> f16 transposed per head: vt[b][h][d][tok]
__global__ void v_transpose_f16_kernel(const float* __restrict__ qkv,
                                       _Float16* __restrict__ vt, int total)
{
  int i = blockIdx.x * blockDim.x + threadIdx.x;
  if (i >= total) return;                 // total = B*N*128
  int c   = i & 127;                      // channel = h*32 + d
  int row = i >> 7;                       // b*N + tok
  int b   = row / NN;
  int tok = row - b*NN;
  int h = c >> 5, d = c & 31;
  float v = qkv[(size_t)row*384 + 2*ENCD + c];
  vt[(((size_t)b*NHEAD + h)*HD + d)*NN + tok] = (_Float16)v;
}

// Writes B*N x 64 (cols 37..63 zero) so GEMM K can be 64.
__global__ void encode_nodes_kernel(const float* __restrict__ st,
                                    const float* __restrict__ pos,
                                    float* __restrict__ Xv, int total)
{
  int r = blockIdx.x * blockDim.x + threadIdx.x;
  if (r >= total) return;
  float* o = Xv + (size_t)r * 64;
#pragma unroll
  for (int i = 0; i < 4; ++i) o[i] = st[(size_t)r*4 + i];
#pragma unroll
  for (int d = 0; d < 3; ++d){
    float p = pos[(size_t)r*3 + d];
#pragma unroll
    for (int i = 0; i < 5; ++i){
      float ang = exp2f((float)i) * PI_F * p;
      o[4 + d*10 + i]     = __cosf(ang);
      o[4 + d*10 + 5 + i] = __sinf(ang);
    }
    o[34 + d] = p;
  }
#pragma unroll
  for (int i = 37; i < 64; ++i) o[i] = 0.0f;
}

__global__ void ft_kernel(const float* __restrict__ time_i,
                          const float* __restrict__ W0, const float* __restrict__ b0,
                          const float* __restrict__ W1, const float* __restrict__ b1,
                          float* __restrict__ outv)
{
  int b = blockIdx.x, t = threadIdx.x;
  float x[11];
  float tv = time_i[b];
#pragma unroll
  for (int i = 0; i < 5; ++i){
    float ang = exp2f((float)i) * PI_F * tv;
    x[i] = __cosf(ang); x[5 + i] = __sinf(ang);
  }
  x[10] = tv;
  float h = b0[t];
#pragma unroll
  for (int k = 0; k < 11; ++k) h += x[k] * W0[k*ENCD + t];
  h = siluf(h);
  __shared__ float hs[ENCD];
  hs[t] = h; __syncthreads();
  float y = b1[t];
  for (int k = 0; k < ENCD; ++k) y += hs[k] * W1[k*ENCD + t];
  outv[b*ENCD + t] = y;
}

__global__ void fc_kernel(const float* __restrict__ cond,
                          const float* __restrict__ W0, const float* __restrict__ b0,
                          const float* __restrict__ W1, const float* __restrict__ b1,
                          float* __restrict__ outv)
{
  int b = blockIdx.x, t = threadIdx.x;
  __shared__ float xs[352];
  for (int j = t; j < 352; j += 128){
    float v;
    if (j < 320){
      int d = j / 10, i = j % 10;
      float p = cond[b*32 + d];
      float ang = exp2f((float)(i < 5 ? i : i - 5)) * PI_F * p;
      v = (i < 5) ? __cosf(ang) : __sinf(ang);
    } else {
      v = cond[b*32 + (j - 320)];
    }
    xs[j] = v;
  }
  __syncthreads();
  float h = b0[t];
  for (int k = 0; k < 352; ++k) h += xs[k] * W0[k*ENCD + t];
  h = siluf(h);
  __shared__ float hs[ENCD];
  hs[t] = h; __syncthreads();
  float y = b1[t];
  for (int k = 0; k < ENCD; ++k) y += hs[k] * W1[k*ENCD + t];
  outv[b*ENCD + t] = y;
}

__global__ void add_rowvec_kernel(float* __restrict__ V,
                                  const float* __restrict__ tv,
                                  const float* __restrict__ cv, int rows)
{
  int i = blockIdx.x * blockDim.x + threadIdx.x;
  if (i >= rows * ENCD) return;
  int row = i >> 7, c = i & 127;
  int b = row / NN;
  V[i] += tv[b*ENCD + c] + cv[b*ENCD + c];
}

// Writes B*NE x 32 (cols 7..31 zero) so GEMM K can be 32.
__global__ void edge_prep_kernel(const int* __restrict__ edges,
                                 const float* __restrict__ pos,
                                 float* __restrict__ Xe,
                                 int* __restrict__ sidx, int* __restrict__ ridx,
                                 float* __restrict__ valid, int total)
{
  int r = blockIdx.x * blockDim.x + threadIdx.x;
  if (r >= total) return;
  int b = r / NEDGE;
  int e0 = edges[2*r], e1 = edges[2*r + 1];
  float vld = (e0 >= 0 && e1 >= 0) ? 1.0f : 0.0f;
  int c0 = min(max(e0, 0), NN - 1);
  int c1 = min(max(e1, 0), NN - 1);
  const float* ps = pos + ((size_t)b*NN + c0) * 3;
  const float* pr = pos + ((size_t)b*NN + c1) * 3;
  float d0 = pr[0]-ps[0], d1 = pr[1]-ps[1], d2 = pr[2]-ps[2];
  float nrm = sqrtf(d0*d0 + d1*d1 + d2*d2);
  float* o = Xe + (size_t)r * 32;
  o[0]=d0; o[1]=d1; o[2]=d2; o[3]=-d0; o[4]=-d1; o[5]=-d2; o[6]=nrm;
#pragma unroll
  for (int i = 7; i < 32; ++i) o[i] = 0.0f;
  sidx[r] = b*NN + c0;
  ridx[r] = b*NN + c1;
  valid[r] = vld;
}

__global__ void mask_rows_kernel(float* __restrict__ x,
                                 const float* __restrict__ mask, int rows)
{
  int i = blockIdx.x * blockDim.x + threadIdx.x;
  if (i >= rows * ENCD) return;
  x[i] *= mask[i >> 7];
}

__global__ void scatter_add_kernel(const float* __restrict__ vals,
                                   const int* __restrict__ ridx,
                                   const float* __restrict__ valid,
                                   float* __restrict__ agg, int erows)
{
  int i = blockIdx.x * blockDim.x + threadIdx.x;
  if (i >= erows * ENCD) return;
  int r = i >> 7, c = i & 127;
  if (valid[r] != 0.0f)
    atomicAdd(&agg[(size_t)ridx[r]*ENCD + c], vals[i]);
}

__global__ void count_kernel(const int* __restrict__ ridx,
                             float* __restrict__ cnt, int erows)
{
  int i = blockIdx.x * blockDim.x + threadIdx.x;
  if (i >= erows) return;
  atomicAdd(&cnt[ridx[i]], 1.0f);   // reference counts ALL edges
}

__global__ void div_mean_kernel(float* __restrict__ agg,
                                const float* __restrict__ cnt, int rows)
{
  int i = blockIdx.x * blockDim.x + threadIdx.x;
  if (i >= rows * ENCD) return;
  agg[i] /= fmaxf(cnt[i >> 7], 1.0f);
}

__global__ void add_kernel(float* __restrict__ dst,
                           const float* __restrict__ src, int n)
{
  int i = blockIdx.x * blockDim.x + threadIdx.x;
  if (i < n) dst[i] += src[i];
}

__global__ void final_kernel(const float* __restrict__ st,
                             const float* __restrict__ pred,
                             float* __restrict__ out, int n)
{
  int i = blockIdx.x * blockDim.x + threadIdx.x;
  if (i < n) out[i] = st[i] + DT_STEP * pred[i];
}

// ---------------------------------------------------------------------------
// Orchestration
// ---------------------------------------------------------------------------
extern "C" void kernel_launch(void* const* d_in, const int* in_sizes, int n_in,
                              void* d_out, int out_size, void* d_ws, size_t ws_size,
                              hipStream_t stream)
{
  (void)in_sizes; (void)n_in; (void)out_size; (void)ws_size;

  const float* state_in = (const float*)d_in[0];
  const float* node_pos = (const float*)d_in[1];
  const int*   edges    = (const int*)  d_in[2];
  const float* time_i   = (const float*)d_in[3];
  const float* cond     = (const float*)d_in[4];
  const float* const* P = (const float* const*)d_in;

  // Param leaf indices (JAX pytree order: dict keys sorted; 'W' < 'b' < 'g'/'l')
  const int I_DEC = 5;   // W0,W1,b0,b1
  const int I_FC  = 9;
  const int I_FE  = 13;
  const int I_FT  = 17;
  const int I_FV  = 21;
  const int I_GNN = 25;  // per block: fe{W0,W1,b0,b1,bt,g} fn{W0,W1,b0,b1,bt,g}
  const int I_TF  = 73;  // per layer: W1,W2,Wo,Wqkv,b1,b2,bo,bqkv,ln1b,ln1g,ln2b,ln2g

  const int MROW = BSZ * NN;       // 6144  (48 * 128)
  const int EROW = BSZ * NEDGE;    // 98304 (768 * 128)

  float* wsp = (float*)d_ws;
  size_t off = 0;
  auto alloc = [&](size_t n) -> float* {
    float* p = wsp + off; off += (n + 63) & ~(size_t)63; return p;
  };
  float* Xv    = alloc((size_t)MROW * 64);     // zero-padded K=64
  float* V     = alloc((size_t)MROW * ENCD);
  float* tvec  = alloc((size_t)BSZ * ENCD);
  float* cvec  = alloc((size_t)BSZ * ENCD);
  float* Xe    = alloc((size_t)EROW * 32);     // zero-padded K=32
  int*   sidx  = (int*)alloc((size_t)EROW);
  int*   ridx  = (int*)alloc((size_t)EROW);
  float* valid = alloc((size_t)EROW);
  float* E     = alloc((size_t)EROW * ENCD);
  float* Ehid  = alloc((size_t)EROW * ENCD);
  float* Enew  = alloc((size_t)EROW * ENCD);
  float* agg   = alloc((size_t)MROW * ENCD);
  float* cnt   = alloc((size_t)MROW);
  float* Nhid  = alloc((size_t)MROW * ENCD);
  float* tmpA  = alloc((size_t)MROW * ENCD);
  float* tmpB  = alloc((size_t)MROW * 2 * ENCD);
  float* QKV   = alloc((size_t)MROW * 3 * ENCD);
  _Float16* QKVh = (_Float16*)alloc((size_t)MROW * 3 * ENCD / 2);  // f16 copy
  _Float16* Vth  = (_Float16*)alloc((size_t)MROW * ENCD / 2);      // f16 V^T
  float* attn  = alloc((size_t)MROW * ENCD);
  float* pred  = alloc((size_t)MROW * 4);

  auto gemm = [&](const float* A, int lda, const int* idx, const float* Wm, int ldw,
                  const float* bias, const float* Cin, float* Cout,
                  int M, int K, int Nc, int act){
    dim3 g((M + GBM - 1)/GBM, (Nc + GBN - 1)/GBN);
    if (idx)
      gemm_wmma_kernel<true ><<<g, 256, 0, stream>>>(A, lda, idx, Wm, ldw, bias, Cin,
                                                     Cout, M, K, Nc, act);
    else
      gemm_wmma_kernel<false><<<g, 256, 0, stream>>>(A, lda, idx, Wm, ldw, bias, Cin,
                                                     Cout, M, K, Nc, act);
  };

  // ---------------- encoders ----------------
  encode_nodes_kernel<<<(MROW + 255)/256, 256, 0, stream>>>(state_in, node_pos, Xv, MROW);
  gemm(Xv, 64, nullptr, P[I_FV], ENCD, P[I_FV+2], nullptr, tmpA, MROW, 64, ENCD, 1);
  gemm(tmpA, ENCD, nullptr, P[I_FV+1], ENCD, P[I_FV+3], nullptr, V, MROW, ENCD, ENCD, 0);
  ft_kernel<<<BSZ, 128, 0, stream>>>(time_i, P[I_FT], P[I_FT+2], P[I_FT+1], P[I_FT+3], tvec);
  fc_kernel<<<BSZ, 128, 0, stream>>>(cond,   P[I_FC], P[I_FC+2], P[I_FC+1], P[I_FC+3], cvec);
  add_rowvec_kernel<<<(MROW*ENCD + 255)/256, 256, 0, stream>>>(V, tvec, cvec, MROW);
  edge_prep_kernel<<<(EROW + 255)/256, 256, 0, stream>>>(edges, node_pos, Xe, sidx, ridx, valid, EROW);
  gemm(Xe, 32, nullptr, P[I_FE], ENCD, P[I_FE+2], nullptr, Ehid, EROW, 32, ENCD, 1);
  gemm(Ehid, ENCD, nullptr, P[I_FE+1], ENCD, P[I_FE+3], nullptr, E, EROW, ENCD, ENCD, 0);

  // ---------------- GNN blocks ----------------
  for (int gb = 0; gb < 4; ++gb){
    const float* feW0 = P[I_GNN + 12*gb + 0];
    const float* feW1 = P[I_GNN + 12*gb + 1];
    const float* feB0 = P[I_GNN + 12*gb + 2];
    const float* feB1 = P[I_GNN + 12*gb + 3];
    const float* feBt = P[I_GNN + 12*gb + 4];
    const float* feG  = P[I_GNN + 12*gb + 5];
    const float* fnW0 = P[I_GNN + 12*gb + 6];
    const float* fnW1 = P[I_GNN + 12*gb + 7];
    const float* fnB0 = P[I_GNN + 12*gb + 8];
    const float* fnB1 = P[I_GNN + 12*gb + 9];
    const float* fnBt = P[I_GNN + 12*gb + 10];
    const float* fnG  = P[I_GNN + 12*gb + 11];

    // edge MLP: concat(S,R,E) @ W0 as 3 accumulating GEMMs (gathered rows)
    gemm(V, ENCD, sidx, feW0,               ENCD, feB0,    nullptr, Ehid, EROW, ENCD, ENCD, 0);
    gemm(V, ENCD, ridx, feW0 + 128*ENCD,    ENCD, nullptr, Ehid,    Ehid, EROW, ENCD, ENCD, 0);
    gemm(E, ENCD, nullptr, feW0 + 256*ENCD, ENCD, nullptr, Ehid,    Ehid, EROW, ENCD, ENCD, 1);
    gemm(Ehid, ENCD, nullptr, feW1, ENCD, feB1, nullptr, Enew, EROW, ENCD, ENCD, 0);
    ln_kernel<<<(EROW + 7)/8, 256, 0, stream>>>(Enew, feG, feBt, Enew, EROW);
    mask_rows_kernel<<<(EROW*ENCD + 255)/256, 256, 0, stream>>>(Enew, valid, EROW);

    // scatter-mean into receiver nodes
    hipMemsetAsync(agg, 0, (size_t)MROW*ENCD*sizeof(float), stream);
    hipMemsetAsync(cnt, 0, (size_t)MROW*sizeof(float), stream);
    scatter_add_kernel<<<(EROW*ENCD + 255)/256, 256, 0, stream>>>(Enew, ridx, valid, agg, EROW);
    count_kernel<<<(EROW + 255)/256, 256, 0, stream>>>(ridx, cnt, EROW);
    div_mean_kernel<<<(MROW*ENCD + 255)/256, 256, 0, stream>>>(agg, cnt, MROW);

    // node MLP: concat(V, agg) @ W0
    gemm(V,   ENCD, nullptr, fnW0,            ENCD, fnB0,    nullptr, Nhid, MROW, ENCD, ENCD, 0);
    gemm(agg, ENCD, nullptr, fnW0 + 128*ENCD, ENCD, nullptr, Nhid,    Nhid, MROW, ENCD, ENCD, 1);
    gemm(Nhid, ENCD, nullptr, fnW1, ENCD, fnB1, nullptr, tmpA, MROW, ENCD, ENCD, 0);
    ln_kernel<<<(MROW + 7)/8, 256, 0, stream>>>(tmpA, fnG, fnBt, tmpA, MROW);

    add_kernel<<<(MROW*ENCD + 255)/256, 256, 0, stream>>>(V, tmpA, MROW*ENCD);
    add_kernel<<<(EROW*ENCD + 255)/256, 256, 0, stream>>>(E, Enew, EROW*ENCD);
  }

  // ---------------- transformer layers ----------------
  for (int tl = 0; tl < 4; ++tl){
    const float* W1   = P[I_TF + 12*tl + 0];
    const float* W2   = P[I_TF + 12*tl + 1];
    const float* Wo   = P[I_TF + 12*tl + 2];
    const float* Wqkv = P[I_TF + 12*tl + 3];
    const float* b1   = P[I_TF + 12*tl + 4];
    const float* b2   = P[I_TF + 12*tl + 5];
    const float* bo   = P[I_TF + 12*tl + 6];
    const float* bqkv = P[I_TF + 12*tl + 7];
    const float* ln1b = P[I_TF + 12*tl + 8];
    const float* ln1g = P[I_TF + 12*tl + 9];
    const float* ln2b = P[I_TF + 12*tl + 10];
    const float* ln2g = P[I_TF + 12*tl + 11];

    ln_kernel<<<(MROW + 7)/8, 256, 0, stream>>>(V, ln1g, ln1b, tmpA, MROW);
    gemm(tmpA, ENCD, nullptr, Wqkv, 3*ENCD, bqkv, nullptr, QKV, MROW, ENCD, 3*ENCD, 0);
    f32_to_f16_kernel<<<(MROW*3*ENCD/4 + 255)/256, 256, 0, stream>>>(QKV, QKVh, MROW*3*ENCD/4);
    v_transpose_f16_kernel<<<(MROW*ENCD + 255)/256, 256, 0, stream>>>(QKV, Vth, MROW*ENCD);
    flash_attn_kernel<<<dim3(NN/64, NHEAD, BSZ), 128, 0, stream>>>(QKVh, Vth, attn);
    gemm(attn, ENCD, nullptr, Wo, ENCD, bo, V, V, MROW, ENCD, ENCD, 0);   // x += mha
    ln_kernel<<<(MROW + 7)/8, 256, 0, stream>>>(V, ln2g, ln2b, tmpA, MROW);
    gemm(tmpA, ENCD, nullptr, W1, 2*ENCD, b1, nullptr, tmpB, MROW, ENCD, 2*ENCD, 2);
    gemm(tmpB, 2*ENCD, nullptr, W2, ENCD, b2, V, V, MROW, 2*ENCD, ENCD, 0); // x += ffn
  }

  // ---------------- decoder ----------------
  gemm(V, ENCD, nullptr, P[I_DEC], ENCD, P[I_DEC+2], nullptr, tmpA, MROW, ENCD, ENCD, 1);
  gemm(tmpA, ENCD, nullptr, P[I_DEC+1], 4, P[I_DEC+3], nullptr, pred, MROW, ENCD, 4, 0);
  final_kernel<<<(MROW*4 + 255)/256, 256, 0, stream>>>(state_in, pred, (float*)d_out, MROW*4);
}
